// SlidingSelfAttention_38946763440805
// MI455X (gfx1250) — matrix operand order repaired and checked by reading, not compiled, not run on hardware
//
#include <hip/hip_runtime.h>
#include <stdint.h>

typedef __attribute__((ext_vector_type(16))) _Float16 v16h;
typedef __attribute__((ext_vector_type(8)))  float    v8f;
typedef __attribute__((ext_vector_type(4)))  uint32_t v4u;
typedef __attribute__((ext_vector_type(4)))  int      v4i;

#define NUM_B 2
#define SEQ_L 2048
#define DIM   1024
#define NHEAD 16
#define DH    64
#define WIN   64
#define PADN  32
#define LPAD  (SEQ_L + 2 * PADN)   // 2112
#define PSTR  98                   // P row stride (fp16 elems)
#define BSTR  72                   // LDS B-tile row stride (fp16 elems) = 144B, 16B-aligned

union FragH { v16h v; uint32_t u[8]; };

__device__ __forceinline__ uint16_t f2h(float f) {
  _Float16 h = (_Float16)f;
  return __builtin_bit_cast(uint16_t, h);
}

__device__ __forceinline__ v8f zero8() {
  v8f z;
#pragma unroll
  for (int i = 0; i < 8; ++i) z[i] = 0.0f;
  return z;
}

// ---- CDNA5 async global->LDS copy (16B per lane), with sync fallback ----
__device__ __forceinline__ void async_cp16(const void* g, void* l) {
#if __has_builtin(__builtin_amdgcn_global_load_async_to_lds_b128)
  typedef __attribute__((address_space(1))) v4i* gp4_t;
  typedef __attribute__((address_space(3))) v4i* lp4_t;
  __builtin_amdgcn_global_load_async_to_lds_b128(
      (gp4_t)(uintptr_t)g, (lp4_t)(uintptr_t)l, 0, 0);
#else
  *(v4u*)l = *(const v4u*)g;
#endif
}

__device__ __forceinline__ void wait_async0() {
#if __has_builtin(__builtin_amdgcn_s_wait_asynccnt)
  __builtin_amdgcn_s_wait_asynccnt(0);
#else
  asm volatile("s_wait_asynccnt 0x0" ::: "memory");
#endif
}

// ---- prep: pad + f32->fp16 convert x into (b, LPAD, DIM) ----
__global__ void pad_convert_x(const float* __restrict__ x, uint16_t* __restrict__ xb) {
  int idx = blockIdx.x * blockDim.x + threadIdx.x;
  const int total = NUM_B * LPAD * DIM;
  if (idx >= total) return;
  int c = idx % DIM;
  int j = (idx / DIM) % LPAD;
  int b = idx / (DIM * LPAD);
  float v = 0.0f;
  if (j >= PADN && j < PADN + SEQ_L)
    v = x[((size_t)b * SEQ_L + (j - PADN)) * DIM + c];
  xb[idx] = f2h(v);
}

// ---- prep: Wt[k][n] = fp16(W[n][k]) ----
__global__ void transpose_w(const float* __restrict__ W, uint16_t* __restrict__ Wt) {
  int idx = blockIdx.x * blockDim.x + threadIdx.x;
  if (idx >= DIM * DIM) return;
  int n = idx % DIM;
  int k = idx / DIM;
  Wt[idx] = f2h(W[(size_t)n * DIM + k]);
}

// ---- fp16 WMMA GEMM: C(MxN) = A(MxK) * B(KxN) [+bias]; B staged in LDS via async ----
// wave -> 32(M) x 64(N); block = 4 waves -> 128x64 tile; double-buffered 32-k chunks
__global__ __launch_bounds__(128) void gemm_f16(
    const uint16_t* __restrict__ A, const uint16_t* __restrict__ Bm,
    void* __restrict__ Cout, const float* __restrict__ bias,
    int M, int N, int K, int outF32) {
  __shared__ uint16_t Bsh[2][32 * BSTR];
  const int lane = threadIdx.x & 31;
  const int wave = threadIdx.x >> 5;
  const int m0 = blockIdx.x * 128 + wave * 32;
  const int n0 = blockIdx.y * 64;
  const int lm = lane & 15;
  const int lh = lane >> 4;
  const uint32_t* A32 = (const uint32_t*)A;
  const int ldA2 = K >> 1;

  // each thread stages two 16B chunks of the 32x64 B tile per k-chunk
  auto issue = [&](int k0, int buf) {
#pragma unroll
    for (int t = 0; t < 2; ++t) {
      int q = t * 128 + (int)threadIdx.x;   // 0..255 chunk id
      int r = q >> 3, c = q & 7;            // row 0..31, 16B-chunk 0..7
      const uint16_t* g = Bm + (size_t)(k0 + r) * N + n0 + c * 8;
      uint16_t* l = &Bsh[buf][r * BSTR + c * 8];
      async_cp16(g, l);
    }
  };

  v8f acc[2][4];
#pragma unroll
  for (int mt = 0; mt < 2; ++mt)
#pragma unroll
    for (int nt = 0; nt < 4; ++nt) acc[mt][nt] = zero8();

  issue(0, 0);
  const int NK = K / 32;
  for (int kk = 0; kk < NK; ++kk) {
    const int buf = kk & 1;
    wait_async0();          // this wave's portion of chunk kk landed
    __syncthreads();        // everyone's portion landed; all reads of buf^1 done
    if (kk + 1 < NK) issue((kk + 1) * 32, buf ^ 1);

    FragH a[2];
#pragma unroll
    for (int mt = 0; mt < 2; ++mt) {
      const uint32_t* ap = A32 + (size_t)(m0 + mt * 16 + lm) * ldA2 + kk * 16 + lh * 4;
#pragma unroll
      for (int i = 0; i < 4; ++i) a[mt].u[i] = ap[i];
#pragma unroll
      for (int i = 0; i < 4; ++i) a[mt].u[4 + i] = ap[8 + i];
    }
    const uint16_t* Bl = &Bsh[buf][lh * 8 * BSTR + lm];   // B col n = lane%16, table-order k
#pragma unroll
    for (int nt = 0; nt < 4; ++nt) {
      FragH b;
#pragma unroll
      for (int i = 0; i < 8; ++i) b.u[i] = (uint32_t)Bl[(2 * i + (i >= 4 ? 8 : 0)) * BSTR + nt * 16] | ((uint32_t)Bl[(2 * i + 1 + (i >= 4 ? 8 : 0)) * BSTR + nt * 16] << 16);
#pragma unroll
      for (int mt = 0; mt < 2; ++mt)
        acc[mt][nt] = __builtin_amdgcn_wmma_f32_16x16x32_f16(
            false, a[mt].v, false, b.v, (short)0, acc[mt][nt], false, false);
    }
  }

  if (outF32) {
    float* C = (float*)Cout;
#pragma unroll
    for (int mt = 0; mt < 2; ++mt)
#pragma unroll
      for (int nt = 0; nt < 4; ++nt) {
        int col = n0 + nt * 16 + lm;
        float bv = bias ? bias[col] : 0.0f;
#pragma unroll
        for (int j = 0; j < 8; ++j) {
          int row = m0 + mt * 16 + j + lh * 8;
          C[(size_t)row * N + col] = acc[mt][nt][j] + bv;
        }
      }
  } else {
    uint16_t* C = (uint16_t*)Cout;
#pragma unroll
    for (int mt = 0; mt < 2; ++mt)
#pragma unroll
      for (int nt = 0; nt < 4; ++nt) {
        int col = n0 + nt * 16 + lm;
#pragma unroll
        for (int j = 0; j < 8; ++j) {
          int row = m0 + mt * 16 + j + lh * 8;
          C[(size_t)row * N + col] = f2h(acc[mt][nt][j]);
        }
      }
  }
}

// ---- sliding-window attention: 1 wave per (b, head, 16-query tile) ----
__global__ __launch_bounds__(128) void attn_sliding(
    const uint16_t* __restrict__ Qp, const uint16_t* __restrict__ Kp,
    const uint16_t* __restrict__ Vp, uint16_t* __restrict__ Ao) {
  __shared__ uint16_t Plds[4][16 * PSTR];
  const int lane = threadIdx.x & 31;
  const int wave = threadIdx.x >> 5;
  const int tid = blockIdx.x * 4 + wave;   // 0..4095
  const int qt = tid & 127;
  const int h  = (tid >> 7) & 15;
  const int bb = tid >> 11;
  const int i0 = qt * 16;
  const int ch = h * DH;
  const int lm = lane & 15;
  const int lh = lane >> 4;

  const uint16_t* Q16 = Qp + (size_t)bb * LPAD * DIM;
  const uint32_t* K32 = (const uint32_t*)(Kp + (size_t)bb * LPAD * DIM);
  const uint32_t* V32 = (const uint32_t*)(Vp + (size_t)bb * LPAD * DIM);

  // B-fragments of Q^T (32d x 16q), strided 16-bit gathers, reused 5x
  FragH qf[2];
#pragma unroll
  for (int c = 0; c < 2; ++c) {
    int d = ch + c * 32 + lh * 8;
#pragma unroll
    for (int i = 0; i < 8; ++i) {
      uint32_t lo = Q16[(size_t)(PADN + i0 + lm) * DIM + d + 2 * i + (i >= 4 ? 8 : 0)];
      uint32_t hi = Q16[(size_t)(PADN + i0 + lm) * DIM + d + 2 * i + 1 + (i >= 4 ? 8 : 0)];
      qf[c].u[i] = lo | (hi << 16);
    }
  }

  // S^T = K_win(80x64) x Q^T
  v8f st[5];
#pragma unroll
  for (int t = 0; t < 5; ++t) {
    v8f accv = zero8();
#pragma unroll
    for (int c = 0; c < 2; ++c) {
      FragH a;
      const uint32_t* ap = K32 + (size_t)(i0 + t * 16 + lm) * (DIM / 2)
                               + ((ch + c * 32) >> 1) + lh * 4;
#pragma unroll
      for (int i = 0; i < 4; ++i) a.u[i] = ap[i];
#pragma unroll
      for (int i = 0; i < 4; ++i) a.u[4 + i] = ap[8 + i];
      accv = __builtin_amdgcn_wmma_f32_16x16x32_f16(
          false, a.v, false, qf[c].v, (short)0, accv, false, false);
    }
    st[t] = accv;
  }

  // band mask (0 <= key-query < 64), scale, softmax per query column
  const float scale = 0.125f;
  float mx = -3.0e38f;
#pragma unroll
  for (int t = 0; t < 5; ++t)
#pragma unroll
    for (int j = 0; j < 8; ++j) {
      int key_rel = t * 16 + j + lh * 8;
      int u = key_rel - lm;
      float v = st[t][j] * scale;
      v = (u >= 0 && u < WIN) ? v : -__builtin_inff();
      st[t][j] = v;
      mx = fmaxf(mx, v);
    }
  mx = fmaxf(mx, __shfl_xor(mx, 16, 32));
  float sum = 0.0f;
#pragma unroll
  for (int t = 0; t < 5; ++t)
#pragma unroll
    for (int j = 0; j < 8; ++j) {
      float e = __expf(st[t][j] - mx);
      st[t][j] = e;
      sum += e;
    }
  sum += __shfl_xor(sum, 16, 32);
  float inv = 1.0f / sum;

  // stage P (16q x 96k, zero-padded) in LDS: D-layout -> A-layout
  uint16_t* P = Plds[wave];
#pragma unroll
  for (int t = 0; t < 5; ++t)
#pragma unroll
    for (int j = 0; j < 8; ++j) {
      int key_rel = t * 16 + j + lh * 8;
      P[lm * PSTR + key_rel] = f2h(st[t][j] * inv);
    }
#pragma unroll
  for (int i = 0; i < 8; ++i)
    P[lm * PSTR + 80 + lh * 8 + i] = 0;

  asm volatile("s_wait_dscnt 0x0" ::: "memory");

  // O = P(16x96) x V(96x64)
  const uint32_t* P32 = (const uint32_t*)P;
#pragma unroll
  for (int nt = 0; nt < 4; ++nt) {
    v8f o = zero8();
#pragma unroll
    for (int kc = 0; kc < 3; ++kc) {
      FragH a, b;
      const uint32_t* ap = P32 + lm * (PSTR / 2) + kc * 16 + lh * 4;
#pragma unroll
      for (int i = 0; i < 4; ++i) a.u[i] = ap[i];
#pragma unroll
      for (int i = 0; i < 4; ++i) a.u[4 + i] = ap[8 + i];
      int key = kc * 32 + lh * 8;
      const uint16_t* bp = Vp + (size_t)bb * LPAD * DIM + ch + nt * 16 + lm;
#pragma unroll
      for (int i = 0; i < 8; ++i) {
        int k0 = key + 2 * i + (i >= 4 ? 8 : 0);
        int r0 = i0 + (k0 < 80 ? k0 : 0), r1 = i0 + (k0 + 1 < 80 ? k0 + 1 : 0);
        b.u[i] = (uint32_t)bp[(size_t)r0 * DIM] | ((uint32_t)bp[(size_t)r1 * DIM] << 16);
      }
      o = __builtin_amdgcn_wmma_f32_16x16x32_f16(
          false, a.v, false, b.v, (short)0, o, false, false);
    }
#pragma unroll
    for (int j = 0; j < 8; ++j) {
      int row = i0 + j + lh * 8;
      Ao[((size_t)bb * SEQ_L + row) * DIM + ch + nt * 16 + lm] = f2h(o[j]);
    }
  }
}

extern "C" void kernel_launch(void* const* d_in, const int* in_sizes, int n_in,
                              void* d_out, int out_size, void* d_ws, size_t ws_size,
                              hipStream_t stream) {
  const float* x  = (const float*)d_in[0];
  const float* Wq = (const float*)d_in[1];
  const float* Wk = (const float*)d_in[2];
  const float* Wv = (const float*)d_in[3];
  const float* Wo = (const float*)d_in[4];
  const float* bo = (const float*)d_in[5];

  uint8_t* ws = (uint8_t*)d_ws;
  size_t off = 0;
  auto alloc = [&](size_t bytes) -> void* {
    void* p = ws + off;
    off = (off + bytes + 255) & ~(size_t)255;
    return p;
  };
  uint16_t* xb  = (uint16_t*)alloc((size_t)NUM_B * LPAD * DIM * 2);
  uint16_t* Wqt = (uint16_t*)alloc((size_t)DIM * DIM * 2);
  uint16_t* Wkt = (uint16_t*)alloc((size_t)DIM * DIM * 2);
  uint16_t* Wvt = (uint16_t*)alloc((size_t)DIM * DIM * 2);
  uint16_t* Wot = (uint16_t*)alloc((size_t)DIM * DIM * 2);
  uint16_t* Qp  = (uint16_t*)alloc((size_t)NUM_B * LPAD * DIM * 2);
  uint16_t* Kp  = (uint16_t*)alloc((size_t)NUM_B * LPAD * DIM * 2);
  uint16_t* Vp  = (uint16_t*)alloc((size_t)NUM_B * LPAD * DIM * 2);
  uint16_t* Ao  = (uint16_t*)alloc((size_t)NUM_B * SEQ_L * DIM * 2);

  {
    int total = NUM_B * LPAD * DIM;
    pad_convert_x<<<(total + 255) / 256, 256, 0, stream>>>(x, xb);
    int tw = DIM * DIM;
    transpose_w<<<(tw + 255) / 256, 256, 0, stream>>>(Wq, Wqt);
    transpose_w<<<(tw + 255) / 256, 256, 0, stream>>>(Wk, Wkt);
    transpose_w<<<(tw + 255) / 256, 256, 0, stream>>>(Wv, Wvt);
    transpose_w<<<(tw + 255) / 256, 256, 0, stream>>>(Wo, Wot);
  }

  dim3 gq((NUM_B * LPAD) / 128, DIM / 64);   // 33 x 16
  gemm_f16<<<gq, 128, 0, stream>>>(xb, Wqt, Qp, nullptr, NUM_B * LPAD, DIM, DIM, 0);
  gemm_f16<<<gq, 128, 0, stream>>>(xb, Wkt, Kp, nullptr, NUM_B * LPAD, DIM, DIM, 0);
  gemm_f16<<<gq, 128, 0, stream>>>(xb, Wvt, Vp, nullptr, NUM_B * LPAD, DIM, DIM, 0);

  attn_sliding<<<(NUM_B * NHEAD * (SEQ_L / 16)) / 4, 128, 0, stream>>>(Qp, Kp, Vp, Ao);

  dim3 go((NUM_B * SEQ_L) / 128, DIM / 64); // 32 x 16
  gemm_f16<<<go, 128, 0, stream>>>(Ao, Wot, d_out, bo, NUM_B * SEQ_L, DIM, DIM, 1);
}
